// NetWrapper_32822140076735
// MI455X (gfx1250) — matrix operand order repaired and checked by reading, not compiled, not run on hardware
//
#include <hip/hip_runtime.h>
#include <stdint.h>

typedef __attribute__((ext_vector_type(16))) __bf16 v16bf;
typedef __attribute__((ext_vector_type(8)))  __bf16 v8bf;
typedef __attribute__((ext_vector_type(8)))  float  v8f;

#define LDSBAR() asm volatile("s_wait_dscnt 0" ::: "memory")

// Workspace offsets (bf16 elements). Each B-operand tile = 32 lanes * 16 bf16 = 512 elems.
#define OFF_W1B 0        // W1   (K=128,N=512): 4*32  tiles = 65536
#define OFF_W2B 65536    // W2   (K=512,N=64) : 16*4  tiles = 32768
#define OFF_W2T 98304    // W2^T (K=64, N=512): 2*32  tiles = 32768
#define OFF_W1T 131072   // W1^T (K=512,N=128): 16*8  tiles = 65536
#define WS_ELEMS 196608

#define E_STEPS 60

// Per-wave LDS partition: sA (16KB) | sG (16KB, reused for dz f32) | sH (2KB) = 34816 B
#define WAVE_LDS 34816
// + block-shared b1 table (512 f32 = 2KB) at the end
#define BLOCK_LDS (8 * WAVE_LDS + 2048)

__device__ __forceinline__ v8f wmma_bf(v16bf a, v16bf b, v8f c) {
  return __builtin_amdgcn_wmma_f32_16x16x32_bf16(false, a, false, b, (short)0, c, false, false);
}

// Pre-swizzled B operand: lane's 16 bf16 are contiguous.
__device__ __forceinline__ v16bf loadB(const __bf16* tile, int lane) {
  return *(const v16bf*)(tile + lane * 16);
}

// Load a 16x32 bf16 A operand (K-tile kt) from an LDS matrix stored in
// C-linear order: elem(row,col) at ((col/16)*8 + row%8)*32 + col%16 + (row>=8?16:0).
__device__ __forceinline__ v16bf loadA_bf(const __bf16* s, int kt, int lane) {
  int rowA = lane & 15, half = (lane >> 4) & 1;
  int c1 = kt * 32 + half * 8, c2 = c1 + 16;
  int i1 = (((c1 >> 4) * 8 + (rowA & 7)) << 5) + (c1 & 15) + ((rowA & 8) << 1);
  int i2 = (((c2 >> 4) * 8 + (rowA & 7)) << 5) + (c2 & 15) + ((rowA & 8) << 1);
  v8bf lo = *(const v8bf*)(s + i1);
  v8bf hi = *(const v8bf*)(s + i2);
  v16bf a;
#pragma unroll
  for (int i = 0; i < 8; ++i) { a[i] = lo[i]; a[8 + i] = hi[i]; }
  return a;
}

__device__ __forceinline__ void st4(float* d, float4 v) { d[0]=v.x; d[1]=v.y; d[2]=v.z; d[3]=v.w; }

// Same addressing as loadA_bf but f32 data (dz staging buffer).
__device__ __forceinline__ void loadA_f32(const float* s, int kt, int lane, float* o) {
  int rowA = lane & 15, half = (lane >> 4) & 1;
  int c1 = kt * 32 + half * 8, c2 = c1 + 16;
  int i1 = (((c1 >> 4) * 8 + (rowA & 7)) << 5) + (c1 & 15) + ((rowA & 8) << 1);
  int i2 = (((c2 >> 4) * 8 + (rowA & 7)) << 5) + (c2 & 15) + ((rowA & 8) << 1);
  st4(o + 0,  *(const float4*)(s + i1));
  st4(o + 4,  *(const float4*)(s + i1 + 4));
  st4(o + 8,  *(const float4*)(s + i2));
  st4(o + 12, *(const float4*)(s + i2 + 4));
}

__device__ __forceinline__ v16bf pack16(const float* f) {
  v16bf v;
#pragma unroll
  for (int i = 0; i < 16; ++i) v[i] = (__bf16)f[i];
  return v;
}

__device__ __forceinline__ uint32_t pkbf(float a, float b) {
  union { __bf16 h[2]; uint32_t u; } x; x.h[0] = (__bf16)a; x.h[1] = (__bf16)b; return x.u;
}
__device__ __forceinline__ float bf_lo(uint32_t u) { union { uint32_t u32; __bf16 h[2]; } x; x.u32 = u; return (float)x.h[0]; }
__device__ __forceinline__ float bf_hi(uint32_t u) { union { uint32_t u32; __bf16 h[2]; } x; x.u32 = u; return (float)x.h[1]; }

// Exact GELU value and derivative (shares the erf term).
__device__ __forceinline__ void gelu_both(float x, float& g, float& dg) {
  float c = 0.5f * (1.0f + erff(x * 0.70710678118654752f));
  float p = 0.39894228040143268f * __expf(-0.5f * x * x);
  g = x * c;
  dg = c + x * p;
}
__device__ __forceinline__ float gelu_grad(float x) {
  float c = 0.5f * (1.0f + erff(x * 0.70710678118654752f));
  return c + x * 0.39894228040143268f * __expf(-0.5f * x * x);
}

// Counter-based hash -> Box-Muller pair (deterministic per (element, step)).
__device__ __forceinline__ uint32_t mixh(uint32_t x) {
  x ^= x >> 17; x *= 0xed5ad4bbu; x ^= x >> 11; x *= 0xac4c1b51u;
  x ^= x >> 15; x *= 0x31848babu; x ^= x >> 14; return x;
}
__device__ __forceinline__ void normal2(uint32_t idx, uint32_t step, float& n0, float& n1) {
  uint32_t s  = mixh(idx * 0x9E3779B9u + step * 0x85EBCA77u + 0x1234567u);
  uint32_t u1 = mixh(s ^ 0x68bc21ebu);
  uint32_t u2 = mixh(s + 0x02e5be93u);
  float f1 = (float)(u1 >> 8) * 5.9604644775390625e-08f + 2.9802322387695312e-08f; // (0,1]
  float f2 = (float)(u2 >> 8) * 5.9604644775390625e-08f;
  float r = __fsqrt_rn(-2.0f * __logf(f1));
  float sn, cs; __sincosf(6.2831853071795865f * f2, &sn, &cs);
  n0 = r * cs; n1 = r * sn;
}

// ---------------- Prep: pack W1, W2, W2^T, W1^T into bf16 B-operand tiles ----------------
__device__ __forceinline__ void pack_one(__bf16* dst, int t, const float* src,
                                         int stride, int Ntiles, bool transpose) {
  int tile = t >> 9, rem = t & 511;
  int ln = rem >> 4, j = rem & 15;
  int nt = tile % Ntiles, kt = tile / Ntiles;
  int k = kt * 32 + ((ln & 16) >> 1) + j + (j & 8);     // lane<16 -> K 0..7,16..23 of tile; lane>=16 -> K 8..15,24..31
  int n = nt * 16 + (ln & 15);
  float v = transpose ? src[n * stride + k] : src[k * stride + n];
  dst[t] = (__bf16)v;
}

__global__ void pack_weights(const float* __restrict__ W1, const float* __restrict__ W2,
                             __bf16* __restrict__ ws) {
  int t = blockIdx.x * 256 + threadIdx.x;
  if (t < 65536)         pack_one(ws + OFF_W1B, t,            W1, 512, 32, false);
  else if (t < 98304)    pack_one(ws + OFF_W2B, t - 65536,    W2, 64,  4,  false);
  else if (t < 131072)   pack_one(ws + OFF_W2T, t - 98304,    W2, 64,  32, true);
  else if (t < WS_ELEMS) pack_one(ws + OFF_W1T, t - 131072,   W1, 512, 8,  true);
}

// ---------------- Fused 60-step Langevin kernel: one wave owns 16 rows ----------------
__global__ void __launch_bounds__(256, 4) langevin_kernel(
    const float* __restrict__ zin, const float* __restrict__ b1p,
    const float* __restrict__ b2p, const float* __restrict__ w3p,
    const __bf16* __restrict__ wsb, float* __restrict__ zout) {
  extern __shared__ __align__(16) char smem_raw[];
  const int lane = threadIdx.x & 31;
  const int wave = threadIdx.x >> 5;
  const int lc   = lane & 15;
  const int half = (lane >> 4) & 1;

  char*     wb  = smem_raw + wave * WAVE_LDS;
  __bf16*   sA  = (__bf16*)wb;              // 16x512 bf16 = 16 KB: a1 / dh1 staging
  uint32_t* sG  = (uint32_t*)(wb + 16384);  // gelu'(h1) packed pairs (16 KB)
  float*    sGz = (float*)(wb + 16384);     // dz f32 staging (8 KB, reuses sG after B1)
  __bf16*   sH  = (__bf16*)(wb + 32768);    // 16x64 bf16 = 2 KB: dh2 staging
  float*    b1s = (float*)(smem_raw + 8 * WAVE_LDS);  // block-shared b1 (512 f32)

  // Cooperative one-time stage of b1 into block LDS.
#pragma unroll
  for (int i = 0; i < 2; ++i) b1s[threadIdx.x + i * 256] = b1p[threadIdx.x + i * 256];
  __syncthreads();

  const int row = blockIdx.x * 128 + wave * 16 + lc;

  const float coefg = -0.5f * 1.0f * 0.4f * 0.4f;   // -alpha*0.5*eps^2
  const float coefp = -0.5f * 1.0f * 0.4f * 0.4f;   // -beta *0.5*eps^2/sig^2
  const float ncoef = 0.4f;                          // gamma*eps

  float b2r[4], w3r[4];
#pragma unroll
  for (int n2 = 0; n2 < 4; ++n2) { b2r[n2] = b2p[n2 * 16 + lc]; w3r[n2] = w3p[n2 * 16 + lc]; }

  // z master copy in registers, A-operand order: zr[kt*16 + j]:
  //   j<8  -> cols kt*32 + half*8 + j ; j>=8 -> cols kt*32 + half*8 + 16 + (j-8)
  float zr[64];
#pragma unroll
  for (int kt = 0; kt < 4; ++kt) {
    int c1 = kt * 32 + half * 8;
    const float* p = zin + (size_t)row * 128 + c1;
    st4(zr + kt * 16 + 0,  *(const float4*)(p + 0));
    st4(zr + kt * 16 + 4,  *(const float4*)(p + 4));
    st4(zr + kt * 16 + 8,  *(const float4*)(p + 16));
    st4(zr + kt * 16 + 12, *(const float4*)(p + 20));
  }

#pragma unroll 1
  for (int it = 0; it < E_STEPS; ++it) {
    // ---- A operands of z (pure register cvt) ----
    v16bf azt[4];
#pragma unroll
    for (int kt = 0; kt < 4; ++kt) azt[kt] = pack16(&zr[kt * 16]);

    // ---- F1: h1 = z@W1 + b1 ; a1 -> sA (bf16), gelu'(h1) packed pairs -> sG ----
#pragma unroll
    for (int nt = 0; nt < 32; ++nt) {
      float bv = b1s[nt * 16 + lc];
      v8f acc = {bv, bv, bv, bv, bv, bv, bv, bv};
#pragma unroll
      for (int kt = 0; kt < 4; ++kt)
        acc = wmma_bf(azt[kt], loadB(wsb + OFF_W1B + (kt * 32 + nt) * 512, lane), acc);
      float dg[8];
#pragma unroll
      for (int e = 0; e < 8; ++e) {
        float g, d; gelu_both(acc[e], g, d);
        sA[(nt * 8 + e) * 32 + lane] = (__bf16)g;
        dg[e] = d;
      }
#pragma unroll
      for (int e = 0; e < 8; e += 2)
        sG[(nt * 4 + (e >> 1)) * 32 + lane] = pkbf(dg[e], dg[e + 1]);
    }
    LDSBAR();

    // ---- F2: h2 = a1@W2 + b2 ----
    v8f acc2[4];
#pragma unroll
    for (int n2 = 0; n2 < 4; ++n2) {
      float bv = b2r[n2];
      acc2[n2] = (v8f){bv, bv, bv, bv, bv, bv, bv, bv};
    }
#pragma unroll
    for (int kt = 0; kt < 16; ++kt) {
      v16bf ak = loadA_bf(sA, kt, lane);
#pragma unroll
      for (int n2 = 0; n2 < 4; ++n2)
        acc2[n2] = wmma_bf(ak, loadB(wsb + OFF_W2B + (kt * 4 + n2) * 512, lane), acc2[n2]);
    }

    // ---- dh2 = gelu'(h2) * W3[col]  -> sH ----
#pragma unroll
    for (int n2 = 0; n2 < 4; ++n2) {
      float w3v = w3r[n2];
#pragma unroll
      for (int e = 0; e < 8; ++e)
        sH[(n2 * 8 + e) * 32 + lane] = (__bf16)(gelu_grad(acc2[n2][e]) * w3v);
    }
    LDSBAR();

    v16bf dA0 = loadA_bf(sH, 0, lane);
    v16bf dA1 = loadA_bf(sH, 1, lane);

    // ---- B1: da1 = dh2@W2^T ; dh1 = da1*gelu'(h1) -> sA (overwrites a1) ----
#pragma unroll
    for (int nt = 0; nt < 32; ++nt) {
      v8f acc = {0.f, 0.f, 0.f, 0.f, 0.f, 0.f, 0.f, 0.f};
      acc = wmma_bf(dA0, loadB(wsb + OFF_W2T + (0 * 32 + nt) * 512, lane), acc);
      acc = wmma_bf(dA1, loadB(wsb + OFF_W2T + (1 * 32 + nt) * 512, lane), acc);
#pragma unroll
      for (int e = 0; e < 8; e += 2) {
        uint32_t pk = sG[(nt * 4 + (e >> 1)) * 32 + lane];
        sA[(nt * 8 + e)     * 32 + lane] = (__bf16)(acc[e]     * bf_lo(pk));
        sA[(nt * 8 + e + 1) * 32 + lane] = (__bf16)(acc[e + 1] * bf_hi(pk));
      }
    }
    LDSBAR();

    // ---- B2: dz = dh1@W1^T, two passes of 4 N-tiles; dz f32 -> sGz (sG is dead) ----
#pragma unroll
    for (int pass = 0; pass < 2; ++pass) {
      v8f acc3[4];
#pragma unroll
      for (int n3 = 0; n3 < 4; ++n3) acc3[n3] = (v8f){0.f, 0.f, 0.f, 0.f, 0.f, 0.f, 0.f, 0.f};
#pragma unroll
      for (int kt = 0; kt < 16; ++kt) {
        v16bf ak = loadA_bf(sA, kt, lane);
#pragma unroll
        for (int n3 = 0; n3 < 4; ++n3)
          acc3[n3] = wmma_bf(ak, loadB(wsb + OFF_W1T + (kt * 8 + pass * 4 + n3) * 512, lane), acc3[n3]);
      }
#pragma unroll
      for (int n3 = 0; n3 < 4; ++n3)
#pragma unroll
        for (int e = 0; e < 8; ++e)
          sGz[((pass * 4 + n3) * 8 + e) * 32 + lane] = acc3[n3][e];
    }
    LDSBAR();

    // ---- update: z += coefg*dz + coefp*z + 0.4*noise ----
#pragma unroll
    for (int kt = 0; kt < 4; ++kt) {
      float dz16[16];
      loadA_f32(sGz, kt, lane, dz16);
      int c1 = kt * 32 + half * 8, c2 = c1 + 16;
#pragma unroll
      for (int j = 0; j < 16; j += 2) {
        int col = (j < 8) ? (c1 + j) : (c2 + (j - 8));
        uint32_t eidx = (uint32_t)(row * 128 + col);
        float n0, n1; normal2(eidx, (uint32_t)it, n0, n1);
        zr[kt * 16 + j]     += coefg * dz16[j]     + coefp * zr[kt * 16 + j]     + ncoef * n0;
        zr[kt * 16 + j + 1] += coefg * dz16[j + 1] + coefp * zr[kt * 16 + j + 1] + ncoef * n1;
      }
    }
    LDSBAR();
  }

  // ---- write z_final ----
#pragma unroll
  for (int kt = 0; kt < 4; ++kt) {
    int c1 = kt * 32 + half * 8;
    float* p = zout + (size_t)row * 128 + c1;
    *(float4*)(p + 0)  = make_float4(zr[kt*16+0],  zr[kt*16+1],  zr[kt*16+2],  zr[kt*16+3]);
    *(float4*)(p + 4)  = make_float4(zr[kt*16+4],  zr[kt*16+5],  zr[kt*16+6],  zr[kt*16+7]);
    *(float4*)(p + 16) = make_float4(zr[kt*16+8],  zr[kt*16+9],  zr[kt*16+10], zr[kt*16+11]);
    *(float4*)(p + 20) = make_float4(zr[kt*16+12], zr[kt*16+13], zr[kt*16+14], zr[kt*16+15]);
  }
}

extern "C" void kernel_launch(void* const* d_in, const int* in_sizes, int n_in,
                              void* d_out, int out_size, void* d_ws, size_t ws_size,
                              hipStream_t stream) {
  const float* z  = (const float*)d_in[0];
  const float* W1 = (const float*)d_in[1];
  const float* b1 = (const float*)d_in[2];
  const float* W2 = (const float*)d_in[3];
  const float* b2 = (const float*)d_in[4];
  const float* W3 = (const float*)d_in[5];
  __bf16* wsb = (__bf16*)d_ws;
  float* out = (float*)d_out;

  // Pack weights into bf16 WMMA B-operand tiles (f32->bf16, pre-swizzled, incl. transposes).
  pack_weights<<<(WS_ELEMS + 255) / 256, 256, 0, stream>>>(W1, W2, wsb);

  const int rows = in_sizes[0] / 128;        // 65536
  const int blocks = rows / 128;             // 8 waves * 16 rows per block
  langevin_kernel<<<blocks, 256, BLOCK_LDS, stream>>>(z, b1, b2, W3, wsb, out);
}
